// HyperbolicJTMPN_11656541241780
// MI455X (gfx1250) — matrix operand order repaired and checked by reading, not compiled, not run on hardware
//
#include <hip/hip_runtime.h>
#include <hip/hip_bf16.h>
#include <stdint.h>

#define HID 128
#define N_TREE 6000
#define N_GRAPH 144000
#define NTOT (N_TREE + N_GRAPH)
#define NEDGE 600000
#define B_SEG 4096
#define ATOM 34
#define EPSV 1e-8f

typedef _Float16 v16h __attribute__((ext_vector_type(16)));
typedef float    v8f  __attribute__((ext_vector_type(8)));

// ---------------- async Global -> LDS staging (CDNA5) ----------------------
// Issues per-lane 16B async copies; caller must later call wait_async_lds()
// and __syncthreads() before reading the LDS region.
template <int LDW>  // f16 elements per global row
__device__ __forceinline__ void async_stage_w(const _Float16* __restrict__ Wg,
                                              _Float16* Wlds, int ldl, int rows) {
  const int chunksPerRow = LDW >> 3;  // 8 f16 = 16B per chunk
  const int total = rows * chunksPerRow;
  for (int c = threadIdx.x; c < total; c += 256) {
    int r  = c / chunksPerRow;
    int cc = c - r * chunksPerRow;
    unsigned lds = (unsigned)(uintptr_t)(Wlds + r * ldl + cc * 8);
    unsigned long long ga = (unsigned long long)(uintptr_t)(Wg + (size_t)r * LDW + cc * 8);
    asm volatile("global_load_async_to_lds_b128 %0, %1, off"
                 :: "v"(lds), "v"(ga) : "memory");
  }
}

__device__ __forceinline__ void wait_async_lds() {
  asm volatile("s_wait_asynccnt 0x0" ::: "memory");
}

// ---------------- WMMA fragment loaders (CDNA5 16x16x32 f16 layouts) -------

// A-matrix 16x32 (MxK), 16-bit: per-half K groups {0..7|16..23} vs {8..15|24..31}
__device__ __forceinline__ v16h frag_a_row(const _Float16* __restrict__ rowp,
                                           int half, int kOff) {
  v16h r;
#pragma unroll
  for (int v = 0; v < 8; ++v) {
    int kb = kOff + ((v >> 2) << 4) + (half << 3) + ((v & 3) << 1);
    const _Float16* p = rowp + kb;
    r[2 * v]     = p[0];
    r[2 * v + 1] = p[1];
  }
  return r;
}

// B-matrix 32x16 (KxN): per-half K groups {0..7|16..23} vs {8..15|24..31} (matches A).
// B[k,n] = W[n_global,k] (y = x @ W^T) -> per lane two 16-byte runs of a W row.
__device__ __forceinline__ v16h frag_b_row(const _Float16* __restrict__ wrow,
                                           int half, int kOff) {
  v16h r;
#pragma unroll
  for (int v = 0; v < 8; ++v) {
    int kb = kOff + ((v >> 2) << 4) + (half << 3) + ((v & 3) << 1);
    const _Float16* p = wrow + kb;
    r[2 * v]     = p[0];
    r[2 * v + 1] = p[1];
  }
  return r;
}

// ---- 128-row tile GEMM + Lorentz nonlinearity, epilogue fully in registers ----
// D fragment: row = v + 8*half, col = n*16 + (lane&15). For a fixed row, all 128
// columns live in the 16 lanes of one half-wave -> shfl_xor reduction width 16.
template <int KSTEPS>
__device__ __forceinline__ void wmma_lorentz(
    const _Float16* __restrict__ Alds, int lda,
    const _Float16* __restrict__ Wlds, int ldw,
    const float* __restrict__ bias, float es,
    float* __restrict__ out, long long rowBase, long long rowLimit) {
  const int lane = threadIdx.x & 31;
  const int wave = threadIdx.x >> 5;
  const int half = lane >> 4;
  const int nl   = lane & 15;
  const int mBase = wave * 16;

  v8f c[8];
#pragma unroll
  for (int n = 0; n < 8; ++n) {
    float bv = bias[n * 16 + nl];
#pragma unroll
    for (int v = 0; v < 8; ++v) c[n][v] = bv;
  }

  const _Float16* arow = Alds + (mBase + nl) * lda;
#pragma unroll
  for (int k = 0; k < KSTEPS; ++k) {
    v16h a = frag_a_row(arow, half, k * 32);
#pragma unroll
    for (int n = 0; n < 8; ++n) {
      v16h b = frag_b_row(Wlds + (n * 16 + nl) * ldw, half, k * 32);
      c[n] = __builtin_amdgcn_wmma_f32_16x16x32_f16(
          false, a, false, b, (short)0, c[n], false, false);
    }
  }

  // Lorentz: time = sigmoid(y0)*e^scale + 1.1 ; rest *= sqrt((t^2-1)/max(Σrest^2,eps))
#pragma unroll
  for (int v = 0; v < 8; ++v) {
    float ss = 0.f;
#pragma unroll
    for (int n = 0; n < 8; ++n) ss += c[n][v] * c[n][v];
#pragma unroll
    for (int off = 1; off < 16; off <<= 1) ss += __shfl_xor(ss, off, 32);
    float y0   = __shfl(c[0][v], half << 4, 32);  // column 0 of this row
    float time = es / (1.f + __expf(-y0)) + 1.1f;
    float rest = ss - y0 * y0;
    float mul  = sqrtf((time * time - 1.f) / fmaxf(rest, EPSV));
    long long row = rowBase + mBase + v + (half << 3);
    if (row < rowLimit) {
      float* orow = out + row * 128;
#pragma unroll
      for (int n = 0; n < 8; ++n) {
        int col = n * 16 + nl;
        orow[col] = (col == 0) ? time : c[n][v] * mul;
      }
    }
  }
}

// ---------------- kernels ----------------

__global__ void cvt_weights(const float* __restrict__ W0, const float* __restrict__ Wl,
                            _Float16* __restrict__ W0h, _Float16* __restrict__ Wlh) {
  int i = blockIdx.x * blockDim.x + threadIdx.x;
  if (i < 128 * 64) {
    int o = i >> 6, k = i & 63;
    W0h[i] = (k < (ATOM + 1)) ? (_Float16)W0[o * (ATOM + 1) + k] : (_Float16)0.f;
  }
  if (i < 3 * 128 * 128) Wlh[i] = (_Float16)Wl[i];
}

__global__ void copy_f32(float* __restrict__ dst, const float* __restrict__ src, long long n) {
  long long i = (long long)blockIdx.x * blockDim.x + threadIdx.x;
  if (i < n) dst[i] = src[i];
}

__global__ void zero_f32v4(float4* __restrict__ p, long long n4) {
  long long i = (long long)blockIdx.x * blockDim.x + threadIdx.x;
  if (i < n4) p[i] = make_float4(0.f, 0.f, 0.f, 0.f);
}

// expmap0 + lorentz_linear(W0): 128 graph rows per block, K padded 35->64
__global__ __launch_bounds__(256) void embed_graph(
    const float* __restrict__ gf, const _Float16* __restrict__ W0h,
    const float* __restrict__ b0, const float* __restrict__ scale0,
    float* __restrict__ X) {
  __shared__ _Float16 Al[128][68];   // 68 f16 stride: 136B rows, conflict-free
  __shared__ _Float16 Wl_s[128][68];
  const int t = threadIdx.x;
  const long long rowBlock = (long long)blockIdx.x * 128;

  async_stage_w<64>(W0h, &Wl_s[0][0], 68, 128);  // overlap with expmap below

  if (t < 128) {
    const float* f = gf + (rowBlock + t) * ATOM;
    float ss = 0.f;
    for (int j = 0; j < ATOM; ++j) { float v = f[j]; ss += v * v; }
    float nrm = fmaxf(sqrtf(ss), EPSV);
    float fac = sinhf(nrm) / nrm;
    Al[t][0] = (_Float16)coshf(nrm);
    for (int j = 0; j < ATOM; ++j) Al[t][1 + j] = (_Float16)(fac * f[j]);
    for (int j = ATOM + 1; j < 64; ++j) Al[t][j] = (_Float16)0.f;
  }
  wait_async_lds();
  __syncthreads();
  float es = __expf(scale0[0]);
  wmma_lorentz<2>(&Al[0][0], 68, &Wl_s[0][0], 68, b0, es,
                  X, (long long)N_TREE + rowBlock, (long long)NTOT);
}

// lorentz_linear(Wl[i]) over all 150000 nodes; optional fused 1/denom scale
// of the input (invd may be null) and optional relu.
__global__ __launch_bounds__(256) void layer_linear(
    const float* __restrict__ Hin, const float* __restrict__ invd,
    const _Float16* __restrict__ Wh, const float* __restrict__ bias,
    const float* __restrict__ scl, int layer, int relu,
    float* __restrict__ Hout) {
  __shared__ _Float16 Al[128][136];   // 272B rows: 16B-aligned chunks, bank 4r+c
  __shared__ _Float16 Wl_s[128][136];
  const int t = threadIdx.x;
  const long long rowBlock = (long long)blockIdx.x * 128;

  async_stage_w<128>(Wh, &Wl_s[0][0], 136, 128);  // async engine runs during cvt

  // speculative prefetch of the next block's A tile (256B per thread)
  if (rowBlock + 128 < NTOT)
    __builtin_prefetch(Hin + (rowBlock + 128) * 128 + (size_t)t * 64, 0, 3);

  for (int idx = t; idx < 128 * 128; idx += 256) {
    int r = idx >> 7, k = idx & 127;
    long long g = rowBlock + r;
    float v = 0.f;
    if (g < NTOT) {
      v = Hin[g * 128 + k];
      if (invd) v *= invd[g];     // invd[g] is wave-uniform -> scalar load
      if (relu) v = fmaxf(v, 0.f);
    }
    Al[r][k] = (_Float16)v;
  }
  wait_async_lds();
  __syncthreads();
  float es = __expf(scl[layer]);
  wmma_lorentz<4>(&Al[0][0], 136, &Wl_s[0][0], 136, bias, es,
                  Hout, rowBlock, (long long)NTOT);
}

// one wave per edge: X[dst] += w * Y[src]   (both buffers ~L2 resident)
__global__ __launch_bounds__(256) void edge_scatter(
    const float* __restrict__ Y, const float* __restrict__ ew,
    const int* __restrict__ esrc, const int* __restrict__ edst,
    float* __restrict__ X) {
  int e = blockIdx.x * (blockDim.x >> 5) + (threadIdx.x >> 5);
  if (e >= NEDGE) return;
  int lane = threadIdx.x & 31;
  int s = esrc[e], d = edst[e];
  float w = ew[e];
  float4 v = ((const float4*)(Y + (size_t)s * 128))[lane];
  float* pd = X + (size_t)d * 128 + lane * 4;
  atomicAdd(pd + 0, w * v.x);
  atomicAdd(pd + 1, w * v.y);
  atomicAdd(pd + 2, w * v.z);
  atomicAdd(pd + 3, w * v.w);
}

// one wave per node: invd[n] = 1/sqrt(max(|x0^2 - Σsp^2|, eps))  (no rewrite of X)
__global__ __launch_bounds__(256) void node_denom(const float* __restrict__ X,
                                                  float* __restrict__ invd, int n) {
  int node = blockIdx.x * (blockDim.x >> 5) + (threadIdx.x >> 5);
  if (node >= n) return;
  int lane = threadIdx.x & 31;
  float4 v = ((const float4*)(X + (size_t)node * 128))[lane];
  float ss = v.x * v.x + v.y * v.y + v.z * v.z + v.w * v.w;
#pragma unroll
  for (int off = 16; off; off >>= 1) ss += __shfl_xor(ss, off, 32);
  float x0 = __shfl(v.x, 0, 32);
  float d = 2.f * x0 * x0 - ss;  // x0^2 - (ss - x0^2)
  if (lane == 0) invd[node] = 1.f / sqrtf(fmaxf(fabsf(d), EPSV));
}

// one wave per graph node: segSum[seg] += h*invd, segCnt[seg] += 1
__global__ __launch_bounds__(256) void seg_scatter(
    const float* __restrict__ X, const float* __restrict__ invd,
    const int* __restrict__ segid,
    float* __restrict__ segSum, float* __restrict__ segCnt) {
  int node = blockIdx.x * (blockDim.x >> 5) + (threadIdx.x >> 5);
  if (node >= N_GRAPH) return;
  int lane = threadIdx.x & 31;
  int seg = segid[node];
  float sc = invd[N_TREE + node];  // wave-uniform
  float4 v = ((const float4*)(X + (size_t)(N_TREE + node) * 128))[lane];
  float* pd = segSum + (size_t)seg * 128 + lane * 4;
  atomicAdd(pd + 0, sc * v.x);
  atomicAdd(pd + 1, sc * v.y);
  atomicAdd(pd + 2, sc * v.z);
  atomicAdd(pd + 3, sc * v.w);
  if (lane == 0) atomicAdd(segCnt + seg, 1.f);
}

// one wave per segment: ave = sum/max(cnt,1); out = ave/centroid_denom(ave)
__global__ __launch_bounds__(256) void seg_finalize(
    const float* __restrict__ segSum, const float* __restrict__ segCnt,
    float* __restrict__ out) {
  int s = blockIdx.x * (blockDim.x >> 5) + (threadIdx.x >> 5);
  if (s >= B_SEG) return;
  int lane = threadIdx.x & 31;
  float inv_c = 1.f / fmaxf(segCnt[s], 1.f);
  float4 v = ((const float4*)(segSum + (size_t)s * 128))[lane];
  v.x *= inv_c; v.y *= inv_c; v.z *= inv_c; v.w *= inv_c;
  float ss = v.x * v.x + v.y * v.y + v.z * v.z + v.w * v.w;
#pragma unroll
  for (int off = 16; off; off >>= 1) ss += __shfl_xor(ss, off, 32);
  float x0 = __shfl(v.x, 0, 32);
  float d = 2.f * x0 * x0 - ss;
  float inv = 1.f / sqrtf(fmaxf(fabsf(d), EPSV));
  float* po = out + (size_t)s * 128 + lane * 4;
  po[0] = v.x * inv; po[1] = v.y * inv; po[2] = v.z * inv; po[3] = v.w * inv;
}

// ---------------- launch ----------------

extern "C" void kernel_launch(void* const* d_in, const int* in_sizes, int n_in,
                              void* d_out, int out_size, void* d_ws, size_t ws_size,
                              hipStream_t stream) {
  (void)in_sizes; (void)n_in; (void)out_size; (void)ws_size;
  const float* gf   = (const float*)d_in[0];
  const float* tf   = (const float*)d_in[1];
  const float* ew   = (const float*)d_in[2];
  const float* W0   = (const float*)d_in[3];
  const float* b0   = (const float*)d_in[4];
  const float* sc0  = (const float*)d_in[5];
  const float* Wl   = (const float*)d_in[6];
  const float* bl   = (const float*)d_in[7];
  const float* scl  = (const float*)d_in[8];
  const int*   esrc = (const int*)d_in[9];
  const int*   edst = (const int*)d_in[10];
  const int*   segi = (const int*)d_in[11];
  float* out = (float*)d_out;

  // workspace: two N x 128 f32 node buffers (~153.6 MB, ~L2 resident on 192MB
  // L2) + f16 weights + per-node 1/denom + segment pooling scratch
  float* X = (float*)d_ws;
  float* Y = X + (size_t)NTOT * 128;
  _Float16* W0h = (_Float16*)(Y + (size_t)NTOT * 128);
  _Float16* Wlh = W0h + 128 * 64;
  float* segSum = (float*)(Wlh + 3 * 128 * 128);
  float* segCnt = segSum + (size_t)B_SEG * 128;
  float* invd   = segCnt + B_SEG;

  cvt_weights<<<192, 256, 0, stream>>>(W0, Wl, W0h, Wlh);
  copy_f32<<<(N_TREE * 128 + 255) / 256, 256, 0, stream>>>(X, tf, (long long)N_TREE * 128);
  embed_graph<<<N_GRAPH / 128, 256, 0, stream>>>(gf, W0h, b0, sc0, X);

  const int gemmBlocks = (NTOT + 127) / 128;
  for (int i = 0; i < 3; ++i) {
    layer_linear<<<gemmBlocks, 256, 0, stream>>>(
        X, (i == 0) ? (const float*)nullptr : invd,
        Wlh + (size_t)i * 128 * 128, bl + i * 128, scl, i, i > 0, Y);
    zero_f32v4<<<((long long)NTOT * 32 + 255) / 256, 256, 0, stream>>>(
        (float4*)X, (long long)NTOT * 32);
    edge_scatter<<<NEDGE / 8, 256, 0, stream>>>(Y, ew, esrc, edst, X);
    node_denom<<<(NTOT + 7) / 8, 256, 0, stream>>>(X, invd, NTOT);
  }

  zero_f32v4<<<((B_SEG * 128 + B_SEG) / 4 + 255) / 256, 256, 0, stream>>>(
      (float4*)segSum, (B_SEG * 128 + B_SEG) / 4);
  seg_scatter<<<N_GRAPH / 8, 256, 0, stream>>>(X, invd, segi, segSum, segCnt);
  seg_finalize<<<B_SEG / 8, 256, 0, stream>>>(segSum, segCnt, out);
}
